// GPT2_32882269618639
// MI455X (gfx1250) — compile-verified, matrix operand order checked by reading, not run on hardware
//
#include <hip/hip_runtime.h>
#include <hip/hip_bf16.h>
#include <math.h>

// ---------------------------------------------------------------------------
// GPT-2 forward for MI455X (gfx1250, wave32, WMMA).
// All dense math runs through v_wmma_f32_16x16x32_bf16 (fp32 accum).
// LDS tiles are stored fragment-major ([m][k] and [n][k], +8 ushort pad) so
// every lane's WMMA fragment is two aligned, bank-conflict-free ds_load_b128.
// ---------------------------------------------------------------------------

typedef __attribute__((ext_vector_type(16))) __bf16 v16bf;
typedef __attribute__((ext_vector_type(2)))  __bf16 v2bf;
typedef __attribute__((ext_vector_type(8)))  float  v8f;

#define LN_EPS_F 1e-5f

// Pack two fp32 -> two bf16 (RNE) in one dword via native conversions.
__device__ __forceinline__ unsigned int pk2bf(float a, float b) {
  v2bf v;
  v[0] = (__bf16)a;
  v[1] = (__bf16)b;
  return __builtin_bit_cast(unsigned int, v);
}
__device__ __forceinline__ float gelu_tanh(float x) {
  const float k0 = 0.7978845608028654f, k1 = 0.044715f;
  float t = tanhf(k0 * (x + k1 * x * x * x));
  return 0.5f * x * (1.0f + t);
}

// Load a 16-element bf16 WMMA fragment from a fragment-major LDS row.
// Element j (0..7) = K hi*8+j, element 8+j = K 16+hi*8+j  (ISA 16-bit layout).
__device__ __forceinline__ v16bf ld_frag(const unsigned short* rowp, int hi) {
  uint4 a = *(const uint4*)(rowp + hi * 8);        // 8 bf16, 16B aligned
  uint4 b = *(const uint4*)(rowp + 16 + hi * 8);   // 8 bf16, 16B aligned
  struct P { uint4 a, b; } p{a, b};
  return __builtin_bit_cast(v16bf, p);
}

// ---------------------------------------------------------------------------
// Tiled WMMA GEMM: C = epilogue(A[M,K] @ B[K,N]); optional B stored [N,K]
// (TRANSB), bias, tanh-GELU, residual add, causal mask, scalar scale.
// Batched over blockIdx.z: offset = (z/zdiv)*strideO + (z%zdiv)*strideI.
// BM=BN=128, BK=32; 256 threads = 8 waves; each wave = 2x4 WMMA 16x16 tiles.
// Requirements (guaranteed by the launcher): M % 128 == 0, K % 32 == 0,
// A rows (and B rows when TRANSB) 16B-aligned. Only N may be ragged.
// ---------------------------------------------------------------------------
template <bool TRANSB, bool GELU, bool CAUSAL>
__global__ __launch_bounds__(256)
void wmma_gemm(const float* __restrict__ A, const float* __restrict__ B,
               const float* __restrict__ bias, const float* __restrict__ resid,
               float* __restrict__ C,
               int M, int N, int K, int lda, int ldb, int ldc,
               int zdiv,
               long long aO, long long aI, long long bO, long long bI,
               long long cO, long long cI, float scale) {
  constexpr int BM = 128, BN = 128, BK = 32, PAD = 8;
  __shared__ unsigned short sA[BM][BK + PAD];   // [m][k] bf16 bits, 80B rows
  __shared__ unsigned short sB[BN][BK + PAD];   // [n][k] bf16 bits, 80B rows

  const int tid  = threadIdx.x;
  const int lane = tid & 31;
  const int warp = tid >> 5;
  const int wm   = warp & 3;     // 4 waves along M
  const int wn   = warp >> 2;    // 2 waves along N
  const int r    = lane & 15;
  const int hi   = lane >> 4;

  const int bm0 = blockIdx.y * BM;
  const int bn0 = blockIdx.x * BN;

  const int z = (int)blockIdx.z;
  const long long zo = z / zdiv, zi = z % zdiv;
  const float* Ab = A + zo * aO + zi * aI;
  const float* Bb = B + zo * bO + zi * bI;
  float*       Cb = C + zo * cO + zi * cI;

  // staging index maps
  const int a_row = tid >> 1;            // 0..127
  const int a_col = (tid & 1) * 16;      // 0 or 16 (K offset)
  const int ng    = (tid & 31) * 4;      // normal-B: n offset 0..124
  const int kg    = (tid >> 5) * 4;      // normal-B: k offset 0..28
  const int tb_n  = tid >> 1;            // trans-B: n 0..127
  const int tb_k  = (tid & 1) * 16;      // trans-B: k 0 or 16

  v8f acc[2][4] = {};

  for (int kt = 0; kt < K; kt += BK) {
    if (kt + BK < K) {
      __builtin_prefetch(&Ab[(size_t)(bm0 + a_row) * lda + kt + BK + a_col], 0, 3);
      if (!TRANSB)
        __builtin_prefetch(&Bb[(size_t)(kt + BK + kg) * ldb + bn0 + ng], 0, 3);
      else
        __builtin_prefetch(&Bb[(size_t)(bn0 + tb_n) * ldb + kt + BK + tb_k], 0, 3);
    }

    // ---- stage A: 16 contiguous fp32 (4x float4) -> 2x uint4 packed bf16
    {
      const float4* src =
          (const float4*)(Ab + (size_t)(bm0 + a_row) * lda + kt + a_col);
      float4 f0 = src[0], f1 = src[1], f2 = src[2], f3 = src[3];
      uint4 lo, hi4;
      lo.x  = pk2bf(f0.x, f0.y); lo.y  = pk2bf(f0.z, f0.w);
      lo.z  = pk2bf(f1.x, f1.y); lo.w  = pk2bf(f1.z, f1.w);
      hi4.x = pk2bf(f2.x, f2.y); hi4.y = pk2bf(f2.z, f2.w);
      hi4.z = pk2bf(f3.x, f3.y); hi4.w = pk2bf(f3.z, f3.w);
      uint4* dst = (uint4*)&sA[a_row][a_col];
      dst[0] = lo; dst[1] = hi4;
    }

    // ---- stage B into [n][k] fragment-major layout
    if (!TRANSB) {
      // 4n x 4k block per thread; only N may be ragged.
      float f[4][4];
      const int gn0 = bn0 + ng;
      if (gn0 + 3 < N) {
#pragma unroll
        for (int kk = 0; kk < 4; ++kk) {
          const float* rp = Bb + (size_t)(kt + kg + kk) * ldb + gn0;
#pragma unroll
          for (int j = 0; j < 4; ++j) f[kk][j] = rp[j];
        }
      } else {
#pragma unroll
        for (int kk = 0; kk < 4; ++kk) {
          const float* rp = Bb + (size_t)(kt + kg + kk) * ldb;
#pragma unroll
          for (int j = 0; j < 4; ++j)
            f[kk][j] = (gn0 + j < N) ? rp[gn0 + j] : 0.0f;
        }
      }
#pragma unroll
      for (int j = 0; j < 4; ++j) {
        uint2 pk;
        pk.x = pk2bf(f[0][j], f[1][j]);
        pk.y = pk2bf(f[2][j], f[3][j]);
        *(uint2*)&sB[ng + j][kg] = pk;
      }
    } else {
      // B stored [N,K]: 16 contiguous fp32 per thread (always aligned here).
      const float4* src =
          (const float4*)(Bb + (size_t)(bn0 + tb_n) * ldb + kt + tb_k);
      float4 f0 = src[0], f1 = src[1], f2 = src[2], f3 = src[3];
      uint4 lo, hi4;
      lo.x  = pk2bf(f0.x, f0.y); lo.y  = pk2bf(f0.z, f0.w);
      lo.z  = pk2bf(f1.x, f1.y); lo.w  = pk2bf(f1.z, f1.w);
      hi4.x = pk2bf(f2.x, f2.y); hi4.y = pk2bf(f2.z, f2.w);
      hi4.z = pk2bf(f3.x, f3.y); hi4.w = pk2bf(f3.z, f3.w);
      uint4* dst = (uint4*)&sB[tb_n][tb_k];
      dst[0] = lo; dst[1] = hi4;
    }
    __syncthreads();

    // ---- fragments: 2x ds_load_b128 each, bank-conflict free
    v16bf af[2];
    v16bf bfr[4];
#pragma unroll
    for (int ti = 0; ti < 2; ++ti)
      af[ti] = ld_frag(&sA[wm * 32 + ti * 16 + r][0], hi);
#pragma unroll
    for (int tj = 0; tj < 4; ++tj)
      bfr[tj] = ld_frag(&sB[wn * 64 + tj * 16 + r][0], hi);

    // ---- 8 WMMAs per wave per K-step
#pragma unroll
    for (int ti = 0; ti < 2; ++ti)
#pragma unroll
      for (int tj = 0; tj < 4; ++tj)
        acc[ti][tj] = __builtin_amdgcn_wmma_f32_16x16x32_bf16(
            false, af[ti], false, bfr[tj], (short)0, acc[ti][tj], false, false);

    __syncthreads();
  }

  // ---- epilogue: scale, bias, GELU, residual, causal mask; N-guard hoisted
#pragma unroll
  for (int ti = 0; ti < 2; ++ti) {
#pragma unroll
    for (int tj = 0; tj < 4; ++tj) {
      const int gn = bn0 + wn * 64 + tj * 16 + r;
      if (gn < N) {
        const float bval = bias ? bias[gn] : 0.0f;
#pragma unroll
        for (int e = 0; e < 8; ++e) {
          const int gm = bm0 + wm * 32 + ti * 16 + hi * 8 + e;
          float v = acc[ti][tj][e] * scale + bval;
          if (GELU)  v = gelu_tanh(v);
          if (resid) v += resid[(size_t)gm * ldc + gn];
          if (CAUSAL && gn > gm) v = -1e30f;
          Cb[(size_t)gm * ldc + gn] = v;
        }
      }
    }
  }
}

// ---------------------------------------------------------------------------
// x[row] = wte[ids[row]] + wpe[row % T]
// ---------------------------------------------------------------------------
__global__ __launch_bounds__(256)
void embed_kernel(const int* __restrict__ ids, const float* __restrict__ wte,
                  const float* __restrict__ wpe, float* __restrict__ x,
                  int T, int E) {
  const int row = blockIdx.x;
  const int t = row % T;
  const int id = ids[row];
  const float* we = wte + (size_t)id * E;
  const float* wp = wpe + (size_t)t * E;
  float* xr = x + (size_t)row * E;
  for (int j = threadIdx.x; j < E; j += 256) xr[j] = we[j] + wp[j];
}

// ---------------------------------------------------------------------------
// LayerNorm, one 256-thread block per row of length E.
// ---------------------------------------------------------------------------
__global__ __launch_bounds__(256)
void ln_kernel(const float* __restrict__ x, const float* __restrict__ w,
               const float* __restrict__ b, float* __restrict__ o, int E) {
  __shared__ float red[256];
  const int row = blockIdx.x, tid = threadIdx.x;
  const float* xr = x + (size_t)row * E;
  float s = 0.f;
  for (int j = tid; j < E; j += 256) s += xr[j];
  red[tid] = s; __syncthreads();
  for (int st = 128; st > 0; st >>= 1) {
    if (tid < st) red[tid] += red[tid + st];
    __syncthreads();
  }
  const float mu = red[0] / E;
  __syncthreads();
  float v = 0.f;
  for (int j = tid; j < E; j += 256) { float d = xr[j] - mu; v += d * d; }
  red[tid] = v; __syncthreads();
  for (int st = 128; st > 0; st >>= 1) {
    if (tid < st) red[tid] += red[tid + st];
    __syncthreads();
  }
  const float rs = rsqrtf(red[0] / E + LN_EPS_F);
  float* orow = o + (size_t)row * E;
  for (int j = tid; j < E; j += 256) orow[j] = (xr[j] - mu) * rs * w[j] + b[j];
}

// ---------------------------------------------------------------------------
// In-place row softmax over T elements (masked entries are -1e30 -> 0).
// ---------------------------------------------------------------------------
__global__ __launch_bounds__(256)
void softmax_kernel(float* __restrict__ p, int T) {
  __shared__ float red[256];
  float* row = p + (size_t)blockIdx.x * T;
  const int tid = threadIdx.x;
  float m = -3.4e38f;
  for (int j = tid; j < T; j += 256) m = fmaxf(m, row[j]);
  red[tid] = m; __syncthreads();
  for (int st = 128; st > 0; st >>= 1) {
    if (tid < st) red[tid] = fmaxf(red[tid], red[tid + st]);
    __syncthreads();
  }
  m = red[0]; __syncthreads();
  float s = 0.f;
  for (int j = tid; j < T; j += 256) {
    float e = __expf(row[j] - m);
    row[j] = e;
    s += e;
  }
  red[tid] = s; __syncthreads();
  for (int st = 128; st > 0; st >>= 1) {
    if (tid < st) red[tid] += red[tid + st];
    __syncthreads();
  }
  const float inv = 1.0f / red[0];
  for (int j = tid; j < T; j += 256) row[j] *= inv;
}

// ---------------------------------------------------------------------------
// Host orchestration.
// ---------------------------------------------------------------------------
extern "C" void kernel_launch(void* const* d_in, const int* in_sizes, int n_in,
                              void* d_out, int out_size, void* d_ws, size_t ws_size,
                              hipStream_t stream) {
  (void)in_sizes; (void)n_in; (void)out_size; (void)ws_size;
  const int L = 11, E = 768, T = 1024, Bn = 4, H = 12, D = 64;
  const int E3 = 3 * E, E4 = 4 * E, V = 50257;
  const int M = Bn * T;  // 4096 token rows

  const int*   ids   = (const int*)d_in[0];
  const float* wte   = (const float*)d_in[1];
  const float* wpe   = (const float*)d_in[2];
  const float* ln1w  = (const float*)d_in[3];
  const float* ln1b  = (const float*)d_in[4];
  const float* attnw = (const float*)d_in[5];
  const float* attnb = (const float*)d_in[6];
  const float* projw = (const float*)d_in[7];
  const float* projb = (const float*)d_in[8];
  const float* ln2w  = (const float*)d_in[9];
  const float* ln2b  = (const float*)d_in[10];
  const float* fcw   = (const float*)d_in[11];
  const float* fcb   = (const float*)d_in[12];
  const float* fpw   = (const float*)d_in[13];
  const float* fpb   = (const float*)d_in[14];
  const float* lnfw  = (const float*)d_in[15];
  const float* lnfb  = (const float*)d_in[16];
  const float* lmw   = (const float*)d_in[17];
  float* out = (float*)d_out;

  // workspace carve-up (256B aligned)
  char* ws = (char*)d_ws;
  size_t off = 0;
  auto carve = [&](size_t bytes) -> float* {
    float* p = (float*)(ws + off);
    off = (off + bytes + 255) & ~(size_t)255;
    return p;
  };
  float* x   = carve((size_t)M * E  * 4);
  float* h   = carve((size_t)M * E  * 4);
  float* qkv = carve((size_t)M * E3 * 4);
  float* y   = carve((size_t)M * E  * 4);
  float* fca = carve((size_t)M * E4 * 4);
  float* sc  = carve((size_t)Bn * H * T * T * 4);

  dim3 blk(256);
  auto grd = [](int m, int n, int z) {
    return dim3((unsigned)((n + 127) / 128), (unsigned)((m + 127) / 128), (unsigned)z);
  };

  embed_kernel<<<M, blk, 0, stream>>>(ids, wte, wpe, x, T, E);

  for (int l = 0; l < L; ++l) {
    const float* aw = attnw + (size_t)l * E * E3;
    const float* ab = attnb + (size_t)l * E3;
    const float* pw = projw + (size_t)l * E * E;
    const float* pb = projb + (size_t)l * E;
    const float* fw = fcw   + (size_t)l * E * E4;
    const float* fb = fcb   + (size_t)l * E4;
    const float* gw = fpw   + (size_t)l * E4 * E;
    const float* gb = fpb   + (size_t)l * E;

    // h = LN1(x)
    ln_kernel<<<M, blk, 0, stream>>>(x, ln1w + (size_t)l * E, ln1b + (size_t)l * E, h, E);

    // qkv = h @ aw + ab                                    [4096,768]x[768,2304]
    wmma_gemm<false, false, false><<<grd(M, E3, 1), blk, 0, stream>>>(
        h, aw, ab, nullptr, qkv, M, E3, E, E, E3, E3,
        1, 0, 0, 0, 0, 0, 0, 1.0f);

    // scores[b,h] = Q Kt / sqrt(D), causal-masked          48 x [1024,64]x[64,1024]
    wmma_gemm<true, false, true><<<grd(T, T, Bn * H), blk, 0, stream>>>(
        qkv, qkv + E, nullptr, nullptr, sc, T, T, D, E3, E3, T,
        H, (long long)T * E3, (long long)D,
           (long long)T * E3, (long long)D,
           (long long)H * T * T, (long long)T * T, 0.125f);

    // softmax rows
    softmax_kernel<<<Bn * H * T, blk, 0, stream>>>(sc, T);

    // y[b,:,h*D:] = P @ V                                  48 x [1024,1024]x[1024,64]
    wmma_gemm<false, false, false><<<grd(T, D, Bn * H), blk, 0, stream>>>(
        sc, qkv + 2 * E, nullptr, nullptr, y, T, D, T, T, E3, E,
        H, (long long)H * T * T, (long long)T * T,
           (long long)T * E3, (long long)D,
           (long long)T * E,  (long long)D, 1.0f);

    // x = x + y @ pw + pb
    wmma_gemm<false, false, false><<<grd(M, E, 1), blk, 0, stream>>>(
        y, pw, pb, x, x, M, E, E, E, E, E,
        1, 0, 0, 0, 0, 0, 0, 1.0f);

    // h = LN2(x)
    ln_kernel<<<M, blk, 0, stream>>>(x, ln2w + (size_t)l * E, ln2b + (size_t)l * E, h, E);

    // fca = gelu(h @ fw + fb)                              [4096,768]x[768,3072]
    wmma_gemm<false, true, false><<<grd(M, E4, 1), blk, 0, stream>>>(
        h, fw, fb, nullptr, fca, M, E4, E, E, E4, E4,
        1, 0, 0, 0, 0, 0, 0, 1.0f);

    // x = x + fca @ gw + gb                                [4096,3072]x[3072,768]
    wmma_gemm<false, false, false><<<grd(M, E, 1), blk, 0, stream>>>(
        fca, gw, gb, x, x, M, E, E4, E4, E, E,
        1, 0, 0, 0, 0, 0, 0, 1.0f);
  }

  // h = LNf(x)
  ln_kernel<<<M, blk, 0, stream>>>(x, lnfw, lnfb, h, E);

  // logits = h @ lm_head_w                                 [4096,768]x[768,50257]
  wmma_gemm<false, false, false><<<grd(M, V, 1), blk, 0, stream>>>(
      h, lmw, nullptr, nullptr, out, M, V, E, E, V, V,
      1, 0, 0, 0, 0, 0, 0, 1.0f);
}
